// DERNN_75642964017600
// MI455X (gfx1250) — matrix operand order repaired and checked by reading, not compiled, not run on hardware
//
#include <hip/hip_runtime.h>
#include <hip/hip_bf16.h>

typedef __attribute__((ext_vector_type(16))) _Float16 v16h;
typedef __attribute__((ext_vector_type(8)))  _Float16 v8h;
typedef __attribute__((ext_vector_type(8)))  float    v8f;

#define NN      16384
#define HD      512
#define INDIM   300
#define KX      320          // x K padded to multiple of 32
#define SX      328          // LDS row strides (halves), padded to kill bank conflicts
#define SH      520
#define SC      1032
#define OFF_X   0
#define OFF_HS  5248         // 16*SX
#define OFF_ES  13568        // OFF_HS + 16*SH
#define LDSH    21888        // OFF_ES + 16*SH  (halves)

__device__ __forceinline__ float sigmoidf(float v) { return 1.0f / (1.0f + __expf(-v)); }

// A-fragment (16x32 f16) from LDS, per ISA layout:
// lane (g = lane>>4, l = lane&15): M = l; halves 0..7 -> K = kb+8g.. , halves 8..15 -> K = kb+16+8g..
__device__ __forceinline__ v16h load_a(const _Float16* p) {
  v8h lo = *(const v8h*)(p);
  v8h hi = *(const v8h*)(p + 16);
  v16h a;
#pragma unroll
  for (int i = 0; i < 8; ++i) { a[i] = lo[i]; a[i + 8] = hi[i]; }
  return a;
}

__device__ __forceinline__ v8f wmma32(v16h a, v16h b, v8f c) {
  return __builtin_amdgcn_wmma_f32_16x16x32_f16(false, a, false, b, (short)0, c, false, false);
}

// K-loop over two output column tiles (n, n+16). b0/b1 are per-lane precomputed
// weight row pointers (B[k][n] = W[n][k]; lane reads halves at +8g and +16+8g, same as A),
// so each iteration is just base + immediate offset.
__device__ __forceinline__ void gemm2(const _Float16* aBase, int K,
                                      const _Float16* __restrict__ b0,
                                      const _Float16* __restrict__ b1,
                                      v8f& c0, v8f& c1) {
#pragma unroll 2
  for (int kb = 0; kb < K; kb += 32) {
    v16h a = load_a(aBase + kb);
    c0 = wmma32(a, load_a(b0 + kb), c0);
    c1 = wmma32(a, load_a(b1 + kb), c1);
  }
}

// ---------------------------------------------------------------------------
// One heap level: block = tile of 16 nodes, 512 threads = 16 wave32s.
// Wave w owns final columns [32w, 32w+32): iu cols n and 512+n, fx n, f n (4 children).
// All accumulators live in registers; LDS only holds A-matrices.
// ---------------------------------------------------------------------------
__global__ __launch_bounds__(512, 1)
void tree_level_kernel(const float* __restrict__ x,
                       const int* __restrict__ child_idx, const int* __restrict__ child_dep,
                       const float* __restrict__ mask_h, const float* __restrict__ mask_e,
                       const float* __restrict__ demb,
                       const _Float16* __restrict__ iux_h, const _Float16* __restrict__ fx_h,
                       const _Float16* __restrict__ iuh_h, const _Float16* __restrict__ iue_h,
                       const _Float16* __restrict__ fh_h,  const _Float16* __restrict__ fe_h,
                       const float* __restrict__ b_iu, const float* __restrict__ b_f,
                       float* __restrict__ H, int base, int count) {
  __shared__ _Float16 sA[LDSH];
  __shared__ int   s_cidx[64];
  __shared__ int   s_cdep[64];
  __shared__ float s_mh[64];
  __shared__ float s_me[64];

  const int tid = threadIdx.x;
  const int bx  = blockIdx.x;
  const int lane = tid & 31;
  const int wid  = tid >> 5;
  const int l = lane & 15;
  const int g = lane >> 4;
  const int nw = wid * 32;

  // ---- meta gather ----
  if (tid < 64) {
    int m = tid >> 2, c = tid & 3;
    bool v = (16 * bx + m) < count;
    size_t t4 = (size_t)(base + 16 * bx + m) * 4 + c;
    s_cidx[tid] = v ? child_idx[t4] : 0;
    s_cdep[tid] = v ? child_dep[t4] : 0;
    s_mh[tid]   = v ? mask_h[t4] : 0.0f;
    s_me[tid]   = v ? mask_e[t4] : 0.0f;
  }
  __syncthreads();

  // ---- Xs gather (f32 -> f16, zero-padded 300..319) ----
  for (int e = tid; e < 16 * KX; e += 512) {
    int m = e / KX, k = e - m * KX;
    bool v = ((16 * bx + m) < count) && (k < INDIM);
    sA[OFF_X + m * SX + k] = v ? (_Float16)x[(size_t)(base + 16 * bx + m) * INDIM + k] : (_Float16)0.0f;
  }
  // ---- h_sum / e_sum gather ----
  for (int e = tid; e < 16 * HD; e += 512) {
    int m = e >> 9, k = e & 511;
    float hs = 0.0f, es = 0.0f;
#pragma unroll
    for (int c = 0; c < 4; ++c) {
      if (s_mh[m * 4 + c] != 0.0f) hs += H[(size_t)s_cidx[m * 4 + c] * HD + k];
      if (s_me[m * 4 + c] != 0.0f) es += demb[(size_t)s_cdep[m * 4 + c] * HD + k];
    }
    sA[OFF_HS + m * SH + k] = (_Float16)hs;
    sA[OFF_ES + m * SH + k] = (_Float16)es;
  }
  __syncthreads();

  const _Float16* aX  = sA + OFF_X  + l * SX + 8 * g;
  const _Float16* aHS = sA + OFF_HS + l * SH + 8 * g;
  const _Float16* aES = sA + OFF_ES + l * SH + 8 * g;
  const size_t rlo = (size_t)(nw + l);        // weight row for column tile n
  const size_t rhi = (size_t)(512 + nw + l);  // weight row for column tile 512+n
  const int go = 8 * g;

  // ---- phase A GEMMs: iu (1024 cols) and fx part of f-preact ----
  v8f ciu0 = {}, ciu1 = {}, ciu2 = {}, ciu3 = {}, cfx0 = {}, cfx1 = {};
  {
    const _Float16* p0 = iux_h + rlo * KX + go;
    gemm2(aX, KX, p0, p0 + (size_t)16 * KX, ciu0, ciu1);
    const _Float16* p1 = iux_h + rhi * KX + go;
    gemm2(aX, KX, p1, p1 + (size_t)16 * KX, ciu2, ciu3);
    const _Float16* p2 = fx_h + rlo * KX + go;
    gemm2(aX, KX, p2, p2 + (size_t)16 * KX, cfx0, cfx1);
  }
  {
    const _Float16* p0 = iuh_h + rlo * HD + go;
    gemm2(aHS, HD, p0, p0 + (size_t)16 * HD, ciu0, ciu1);
    const _Float16* p1 = iuh_h + rhi * HD + go;
    gemm2(aHS, HD, p1, p1 + (size_t)16 * HD, ciu2, ciu3);
  }
  {
    const _Float16* p0 = iue_h + rlo * HD + go;
    gemm2(aES, HD, p0, p0 + (size_t)16 * HD, ciu0, ciu1);
    const _Float16* p1 = iue_h + rhi * HD + go;
    gemm2(aES, HD, p1, p1 + (size_t)16 * HD, ciu2, ciu3);
  }

  // ---- phase B: per-child forget gates, fa += sigmoid(fpre) * ch ----
  v8f fa0 = {}, fa1 = {};
  const _Float16* aC = sA + l * SC + 8 * g;     // CE overlay at offset 0 (barrier-protected)
  const _Float16* pFH = fh_h + rlo * HD + go;
  const _Float16* pFE = fe_h + rlo * HD + go;
#pragma unroll 1
  for (int c = 0; c < 4; ++c) {
    __syncthreads();                             // everyone done reading prev LDS contents
    for (int e = tid; e < 16 * 1024; e += 512) { // CE = [ch_c | e_c], f16
      int m = e >> 10, k = e & 1023;
      float v;
      if (k < 512) v = (s_mh[m * 4 + c] != 0.0f) ? H[(size_t)s_cidx[m * 4 + c] * HD + k] : 0.0f;
      else         v = demb[(size_t)s_cdep[m * 4 + c] * HD + (k - 512)];
      sA[m * SC + k] = (_Float16)v;
    }
    __syncthreads();

    v8f cf0 = cfx0, cf1 = cfx1;
    gemm2(aC,       HD, pFH, pFH + (size_t)16 * HD, cf0, cf1);   // ch @ fh_w^T
    gemm2(aC + 512, HD, pFE, pFE + (size_t)16 * HD, cf0, cf1);   // e  @ fe_w^T

    float bf0 = b_f[nw + l], bf1 = b_f[nw + 16 + l];
#pragma unroll
    for (int r = 0; r < 8; ++r) {                // C/D layout: M = r + 8g, N = nbase + l
      int m = r + 8 * g;
      fa0[r] += sigmoidf(cf0[r] + bf0) * (float)sA[m * SC + nw + l];
      fa1[r] += sigmoidf(cf1[r] + bf1) * (float)sA[m * SC + nw + 16 + l];
    }
  }

  // ---- finalize: h = tanh(sigmoid(iu_i)*tanh(iu_u) + fa); write H ----
  const int tb = base + 16 * bx;
  float bi0 = b_iu[nw + l],       bi1 = b_iu[nw + 16 + l];
  float bu0 = b_iu[512 + nw + l], bu1 = b_iu[512 + nw + 16 + l];
#pragma unroll
  for (int r = 0; r < 8; ++r) {
    int m = r + 8 * g;
    if (16 * bx + m < count) {
      float h0 = tanhf(sigmoidf(ciu0[r] + bi0) * tanhf(ciu2[r] + bu0) + fa0[r]);
      float h1 = tanhf(sigmoidf(ciu1[r] + bi1) * tanhf(ciu3[r] + bu1) + fa1[r]);
      H[(size_t)(tb + m) * HD + nw + l]      = h0;
      H[(size_t)(tb + m) * HD + nw + 16 + l] = h1;
    }
  }
}

// f32 -> f16 weight cast with row zero-padding (src rows scols wide -> dst rows dcols wide)
__global__ void cvt_f16_kernel(_Float16* __restrict__ dst, const float* __restrict__ src,
                               int rows, int scols, int dcols) {
  int gidx = blockIdx.x * blockDim.x + threadIdx.x;
  if (gidx >= rows * dcols) return;
  int r = gidx / dcols, k = gidx - r * dcols;
  dst[gidx] = (k < scols) ? (_Float16)src[(size_t)r * scols + k] : (_Float16)0.0f;
}

__global__ void bias_comb_kernel(float* __restrict__ b_iu, float* __restrict__ b_f,
                                 const float* __restrict__ iux_b, const float* __restrict__ iuh_b,
                                 const float* __restrict__ iue_b, const float* __restrict__ fx_b,
                                 const float* __restrict__ fh_b,  const float* __restrict__ fe_b) {
  int gidx = blockIdx.x * blockDim.x + threadIdx.x;
  if (gidx < 1024) b_iu[gidx] = iux_b[gidx] + iuh_b[gidx] + iue_b[gidx];
  if (gidx < 512)  b_f[gidx]  = fx_b[gidx] + fh_b[gidx] + fe_b[gidx];
}

__global__ void root_copy_kernel(float* __restrict__ out, const float* __restrict__ H) {
  int k = threadIdx.x;
  out[k] = H[(size_t)(NN - 1) * HD + k];
}

extern "C" void kernel_launch(void* const* d_in, const int* in_sizes, int n_in,
                              void* d_out, int out_size, void* d_ws, size_t ws_size,
                              hipStream_t stream) {
  (void)in_sizes; (void)n_in; (void)out_size; (void)ws_size;
  const float* x         = (const float*)d_in[0];
  const int*   child_idx = (const int*)d_in[1];
  const int*   child_dep = (const int*)d_in[2];
  const float* mask_h    = (const float*)d_in[3];
  const float* mask_e    = (const float*)d_in[4];
  const float* demb      = (const float*)d_in[5];
  const float* iux_w = (const float*)d_in[6];  const float* iux_b = (const float*)d_in[7];
  const float* iuh_w = (const float*)d_in[8];  const float* iuh_b = (const float*)d_in[9];
  const float* iue_w = (const float*)d_in[10]; const float* iue_b = (const float*)d_in[11];
  const float* fx_w  = (const float*)d_in[12]; const float* fx_b  = (const float*)d_in[13];
  const float* fh_w  = (const float*)d_in[14]; const float* fh_b  = (const float*)d_in[15];
  const float* fe_w  = (const float*)d_in[16]; const float* fe_b  = (const float*)d_in[17];

  char* ws = (char*)d_ws;
  float*    H      = (float*)ws;                                  // 16384*512*4 = 33554432
  _Float16* iux_h  = (_Float16*)(ws + 33554432);                  // 1024*320*2
  _Float16* fx_h   = (_Float16*)(ws + 34209792);                  // 512*320*2
  _Float16* iuh_h  = (_Float16*)(ws + 34537472);                  // 1024*512*2
  _Float16* iue_h  = (_Float16*)(ws + 35586048);                  // 1024*512*2
  _Float16* fh_h   = (_Float16*)(ws + 36634624);                  // 512*512*2
  _Float16* fe_h   = (_Float16*)(ws + 37158912);                  // 512*512*2
  float*    b_iu   = (float*)(ws + 37683200);                     // 1024*4
  float*    b_f    = (float*)(ws + 37687296);                     // 512*4

  cvt_f16_kernel<<<(1024 * KX + 255) / 256, 256, 0, stream>>>(iux_h, iux_w, 1024, 300, KX);
  cvt_f16_kernel<<<(512  * KX + 255) / 256, 256, 0, stream>>>(fx_h,  fx_w,  512,  300, KX);
  cvt_f16_kernel<<<(1024 * HD + 255) / 256, 256, 0, stream>>>(iuh_h, iuh_w, 1024, 512, HD);
  cvt_f16_kernel<<<(1024 * HD + 255) / 256, 256, 0, stream>>>(iue_h, iue_w, 1024, 512, HD);
  cvt_f16_kernel<<<(512  * HD + 255) / 256, 256, 0, stream>>>(fh_h,  fh_w,  512,  512, HD);
  cvt_f16_kernel<<<(512  * HD + 255) / 256, 256, 0, stream>>>(fe_h,  fe_w,  512,  512, HD);
  bias_comb_kernel<<<4, 256, 0, stream>>>(b_iu, b_f, iux_b, iuh_b, iue_b, fx_b, fh_b, fe_b);

  // 4-ary heap levels, leaves first (node-index ranges; children are exactly one level earlier)
  static const int lstart[8] = {0, 10923, 15019, 16043, 16299, 16363, 16379, 16383};
  static const int lcount[8] = {10923, 4096, 1024, 256, 64, 16, 4, 1};
  for (int L = 0; L < 8; ++L) {
    tree_level_kernel<<<(lcount[L] + 15) / 16, 512, 0, stream>>>(
        x, child_idx, child_dep, mask_h, mask_e, demb,
        iux_h, fx_h, iuh_h, iue_h, fh_h, fe_h, b_iu, b_f, H, lstart[L], lcount[L]);
  }
  root_copy_kernel<<<1, 512, 0, stream>>>((float*)d_out, H);
}
